// GatedDeltaNet_83571473645758
// MI455X (gfx1250) — compile-verified, matrix operand order checked by reading, not run on hardware
//
#include <hip/hip_runtime.h>
#include <math.h>

typedef __bf16 bf16_t;
typedef __bf16 v16bf __attribute__((ext_vector_type(16)));
typedef float  v8f   __attribute__((ext_vector_type(8)));
typedef unsigned int u32x4 __attribute__((ext_vector_type(4)));
typedef int i32x8 __attribute__((ext_vector_type(8)));
typedef int i32x4 __attribute__((ext_vector_type(4)));

#define WMMA_BF16(A_, B_, C_) \
  __builtin_amdgcn_wmma_f32_16x16x32_bf16(false, (A_), false, (B_), (short)0, (C_), false, false)

// ---------------- problem constants ----------------
static constexpr int kB    = 2;
static constexpr int kT    = 2048;
static constexpr int kHID  = 2048;
static constexpr int kNH   = 16;
static constexpr int kD    = 128;
static constexpr int kQKVN = 6144;
static constexpr int kABN  = 2080;
static constexpr int kBT   = kB * kT;
static constexpr int kCH   = 64;   // delta chunk length
static constexpr int kNS   = 4;    // val-dim splits per head (occupancy)
static constexpr int kSL   = kD / kNS;  // 32 val channels per split

// ---------------- Tensor Data Mover: 2-D tile (bf16) -> LDS ----------------
__device__ inline unsigned lds_off(const void* p) {
  return (unsigned)(unsigned long long)p;  // LDS flat addr low 32 bits = offset
}

__device__ inline void tdm_load_2d(const void* gptr, unsigned ldsoff,
                                   unsigned rows, unsigned rowlen, unsigned strideElems) {
  unsigned long long ga = (unsigned long long)gptr;
  u32x4 g0 = { 1u, ldsoff, (unsigned)ga,
               (unsigned)((ga >> 32) & 0x1FFFFFFull) | 0x80000000u };
  i32x8 g1 = { (int)0x00010000u,          // data_size = 2 bytes
               (int)(rowlen << 16),       // tensor_dim0[15:0]
               (int)(rows << 16),         // tensor_dim1[15:0]
               (int)(rowlen << 16),       // tile_dim0
               (int)rows,                 // tile_dim1
               (int)strideElems,          // tensor_dim0_stride[31:0]
               0, 0 };
  i32x4 z4 = { 0, 0, 0, 0 };
  i32x8 z8 = { 0, 0, 0, 0, 0, 0, 0, 0 };
  __builtin_amdgcn_tensor_load_to_lds(g0, g1, z4, z4, z8, 0);
}

// ---------------- fast elementwise math (hardware transcendentals) ---------
__device__ inline float fexp(float x) { return __expf(x); }
__device__ inline float fsig(float x) { return 1.f / (1.f + __expf(-x)); }

// ---------------- WMMA fragment helpers (wave32, 16x16x32 bf16) ------------
__device__ inline int frag_kbase(int v, int lane) {
  return ((v >= 4) ? 16 : 0) + ((v & 3) << 1) + ((lane >> 4) << 3);
}

// A-fragment (row-major src): contiguous per lane -> ds_load_b128 pairs.
// Also used for B operands stored TRANSPOSED (row index = output column).
__device__ inline v16bf ldA_bf(const bf16_t* p, int ld, int row0, int k0) {
  int lane = threadIdx.x & 31;
  const bf16_t* base = p + (size_t)(row0 + (lane & 15)) * ld + k0;
  v16bf f;
#pragma unroll
  for (int v = 0; v < 8; ++v) {
    int kb = frag_kbase(v, lane);
    f[2 * v]     = base[kb];
    f[2 * v + 1] = base[kb + 1];
  }
  return f;
}

__device__ inline v16bf ldA_bf_rowscale(const bf16_t* p, int ld, int row0, int k0,
                                        const float* srow) {
  int lane = threadIdx.x & 31;
  int r = row0 + (lane & 15);
  float sc = srow[r];
  const bf16_t* base = p + (size_t)r * ld + k0;
  v16bf f;
#pragma unroll
  for (int v = 0; v < 8; ++v) {
    int kb = frag_kbase(v, lane);
    f[2 * v]     = (bf16_t)((float)base[kb] * sc);
    f[2 * v + 1] = (bf16_t)((float)base[kb + 1] * sc);
  }
  return f;
}

// B = X^T with X row-major: element (k,c) = X[c][k] -> contiguous per lane
__device__ inline v16bf ldBT_bf(const bf16_t* p, int ld, int k0, int col0) {
  int lane = threadIdx.x & 31;
  const bf16_t* rowp = p + (size_t)(col0 + (lane & 15)) * ld;
  v16bf f;
#pragma unroll
  for (int v = 0; v < 8; ++v) {
    int kb = k0 + frag_kbase(v, lane);
    f[2 * v]     = rowp[kb];
    f[2 * v + 1] = rowp[kb + 1];
  }
  return f;
}

__device__ inline v8f ldC_f32_scaled(const float* p, int ld, int row0, int col0, float sc) {
  int lane = threadIdx.x & 31;
  int c = col0 + (lane & 15);
  int r = row0 + ((lane >> 4) << 3);
  v8f acc;
#pragma unroll
  for (int v = 0; v < 8; ++v) acc[v] = p[(size_t)(r + v) * ld + c] * sc;
  return acc;
}

__device__ inline v8f ldC_bf16(const bf16_t* p, int ld, int row0, int col0) {
  int lane = threadIdx.x & 31;
  int c = col0 + (lane & 15);
  int r = row0 + ((lane >> 4) << 3);
  v8f acc;
#pragma unroll
  for (int v = 0; v < 8; ++v) acc[v] = (float)p[(size_t)(r + v) * ld + c];
  return acc;
}

__device__ inline void stC_f32(float* p, int ld, int row0, int col0, v8f acc) {
  int lane = threadIdx.x & 31;
  int c = col0 + (lane & 15);
  int r = row0 + ((lane >> 4) << 3);
#pragma unroll
  for (int v = 0; v < 8; ++v) p[(size_t)(r + v) * ld + c] = acc[v];
}

__device__ inline void stC_bf16(bf16_t* p, int ld, int row0, int col0, v8f acc) {
  int lane = threadIdx.x & 31;
  int c = col0 + (lane & 15);
  int r = row0 + ((lane >> 4) << 3);
#pragma unroll
  for (int v = 0; v < 8; ++v) p[(size_t)(r + v) * ld + c] = (bf16_t)acc[v];
}

__device__ inline unsigned short bf_bits(bf16_t x) {
  return __builtin_bit_cast(unsigned short, x);
}

// Transposed C store: element (r+v, c) -> p[c*ld + r+v]; contiguous over v ->
// one packed ds_store_b128.
__device__ inline void stC_bf16_T(bf16_t* p, int ld, int row0, int col0, v8f acc) {
  int lane = threadIdx.x & 31;
  int c = col0 + (lane & 15);
  int r = row0 + ((lane >> 4) << 3);
  unsigned w[4];
#pragma unroll
  for (int j = 0; j < 4; ++j) {
    w[j] = (unsigned)bf_bits((bf16_t)acc[2 * j]) |
           ((unsigned)bf_bits((bf16_t)acc[2 * j + 1]) << 16);
  }
  int4 pk; pk.x = (int)w[0]; pk.y = (int)w[1]; pk.z = (int)w[2]; pk.w = (int)w[3];
  *(int4*)(p + (size_t)c * ld + r) = pk;
}

// ---------------- f32 -> bf16 conversion ----------------
__global__ __launch_bounds__(256) void cvt_bf16_kernel(const float* __restrict__ in,
                                                       bf16_t* __restrict__ out, size_t n) {
  size_t i = (size_t)blockIdx.x * blockDim.x + threadIdx.x;
  if (i < n) out[i] = (bf16_t)in[i];
}

// ---------------- bf16 GEMM with double-buffered TDM staging ---------------
__global__ __launch_bounds__(256) void gemm_bf16_kernel(const bf16_t* __restrict__ A,
                                                        const bf16_t* __restrict__ B,
                                                        float* __restrict__ C,
                                                        int M, int N, int K) {
  __shared__ bf16_t sA[2][64 * 64];    // row-major [m][k]
  __shared__ bf16_t sBt[2][64 * 64];   // transposed [n][k]
  int col0 = blockIdx.x * 64;
  int row0 = blockIdx.y * 64;
  int tid = threadIdx.x;
  int wave = tid >> 5;

  int t0 = wave, t1 = wave + 8;
  int tr0 = (t0 >> 2) * 16, tc0 = (t0 & 3) * 16;
  int tr1 = (t1 >> 2) * 16, tc1 = (t1 & 3) * 16;
  v8f acc0 = {}; v8f acc1 = {};

  const bool bstage = tid < 128;
  int pair = tid >> 2, cseg = (tid & 3) * 16, k0r = 2 * pair;
  const bool bvalid = (col0 + cseg) < N;   // N % 32 == 0

  // prologue: stage buffer 0 (and permanently zero OOB columns of both bufs)
  if (wave == 4) tdm_load_2d(A + (size_t)row0 * K, lds_off(sA[0]), 64, 64, (unsigned)K);
  if (bstage) {
    if (bvalid) {
      union { int4 v[2]; unsigned short u[16]; } r0, r1;
      const int4* g0 = (const int4*)(B + (size_t)k0r * N + col0 + cseg);
      const int4* g1 = (const int4*)(B + (size_t)(k0r + 1) * N + col0 + cseg);
      r0.v[0] = g0[0]; r0.v[1] = g0[1];
      r1.v[0] = g1[0]; r1.v[1] = g1[1];
#pragma unroll
      for (int c = 0; c < 16; ++c) {
        unsigned pk = (unsigned)r0.u[c] | ((unsigned)r1.u[c] << 16);
        *(unsigned*)(sBt[0] + (cseg + c) * 64 + k0r) = pk;
      }
    } else {
#pragma unroll
      for (int c = 0; c < 16; ++c) {
        *(unsigned*)(sBt[0] + (cseg + c) * 64 + k0r) = 0u;
        *(unsigned*)(sBt[1] + (cseg + c) * 64 + k0r) = 0u;
      }
    }
  }
  __builtin_amdgcn_s_wait_tensorcnt(0);
  __syncthreads();

  for (int kk = 0; kk < K; kk += 64) {
    int cur = (kk >> 6) & 1;
    int nxt = cur ^ 1;
    bool havenext = (kk + 64) < K;

    // issue next-tile TDM + next-B global loads before compute (latency hide)
    union { int4 v[2]; unsigned short u[16]; } n0, n1;
    if (havenext) {
      if (wave == 4) {
        tdm_load_2d(A + (size_t)row0 * K + kk + 64, lds_off(sA[nxt]), 64, 64, (unsigned)K);
      }
      if (bstage && bvalid) {
        const int4* g0 = (const int4*)(B + (size_t)(kk + 64 + k0r) * N + col0 + cseg);
        const int4* g1 = (const int4*)(B + (size_t)(kk + 64 + k0r + 1) * N + col0 + cseg);
        n0.v[0] = g0[0]; n0.v[1] = g0[1];
        n1.v[0] = g1[0]; n1.v[1] = g1[1];
      }
    }
#pragma unroll
    for (int k2 = 0; k2 < 64; k2 += 32) {
      v16bf a0 = ldA_bf(sA[cur], 64, tr0, k2);
      v16bf b0 = ldA_bf(sBt[cur], 64, tc0, k2);
      acc0 = WMMA_BF16(a0, b0, acc0);
      v16bf a1 = ldA_bf(sA[cur], 64, tr1, k2);
      v16bf b1 = ldA_bf(sBt[cur], 64, tc1, k2);
      acc1 = WMMA_BF16(a1, b1, acc1);
    }
    if (havenext && bstage && bvalid) {
#pragma unroll
      for (int c = 0; c < 16; ++c) {
        unsigned pk = (unsigned)n0.u[c] | ((unsigned)n1.u[c] << 16);
        *(unsigned*)(sBt[nxt] + (cseg + c) * 64 + k0r) = pk;
      }
    }
    __builtin_amdgcn_s_wait_tensorcnt(0);
    __syncthreads();
  }
  if (col0 + tc0 < N) stC_f32(C + (size_t)row0 * N + col0, N, tr0, tc0, acc0);
  if (col0 + tc1 < N) stC_f32(C + (size_t)row0 * N + col0, N, tr1, tc1, acc1);
}

// ---------------- conv(K=4) + SiLU + q/k L2 norm + bf16 repack -------------
__global__ __launch_bounds__(256) void conv_silu_prep_kernel(
    const float* __restrict__ qkv, const float* __restrict__ cw,
    bf16_t* __restrict__ qn, bf16_t* __restrict__ kn, bf16_t* __restrict__ vn) {
  int gw = (blockIdx.x * blockDim.x + threadIdx.x) >> 5;
  int lane = threadIdx.x & 31;
  if (gw >= kBT * kNH) return;
  int bt = gw >> 4;
  int h = gw & 15;
  int tloc = bt % kT;

  const float rscale = 0.08838834764831845f;  // 128^-0.5
  float vq[4], vk[4], vv[4];
  float ssq = 0.f, ssk = 0.f;
#pragma unroll
  for (int i = 0; i < 4; ++i) {
    int c = lane + 32 * i;
    int cq = h * kD + c;
    int ck = 2048 + h * kD + c;
    int cv = 4096 + h * kD + c;
    float aq = 0.f, ak = 0.f, av = 0.f;
#pragma unroll
    for (int j = 0; j < 4; ++j) {
      if (tloc + j >= 3) {
        const float* rp = qkv + (size_t)(bt + j - 3) * kQKVN;
        aq += cw[cq * 4 + j] * rp[cq];
        ak += cw[ck * 4 + j] * rp[ck];
        av += cw[cv * 4 + j] * rp[cv];
      }
    }
    aq *= fsig(aq);
    ak *= fsig(ak);
    av *= fsig(av);
    vq[i] = aq; vk[i] = ak; vv[i] = av;
    ssq += aq * aq; ssk += ak * ak;
  }
#pragma unroll
  for (int off = 16; off; off >>= 1) {
    ssq += __shfl_xor(ssq, off);
    ssk += __shfl_xor(ssk, off);
  }
  float nq = rsqrtf(ssq + 1e-6f) * rscale;
  float nk = rsqrtf(ssk + 1e-6f);
#pragma unroll
  for (int i = 0; i < 4; ++i) {
    int c = lane + 32 * i;
    size_t o = (size_t)bt * kHID + h * kD + c;
    qn[o] = (bf16_t)(vq[i] * nq);
    kn[o] = (bf16_t)(vk[i] * nk);
    vn[o] = (bf16_t)vv[i];
  }
}

// ---------------- g / beta ----------------
__global__ __launch_bounds__(256) void gbeta_kernel(const float* __restrict__ ab,
                                                    const float* __restrict__ A_log,
                                                    const float* __restrict__ dt_bias,
                                                    float* __restrict__ g,
                                                    float* __restrict__ beta) {
  int i = blockIdx.x * blockDim.x + threadIdx.x;
  if (i >= kBT * kNH) return;
  int bt = i >> 4;
  int h = i & 15;
  float a = ab[(size_t)bt * kABN + h] + dt_bias[h];
  float sp = (a > 20.f) ? a : __logf(1.f + __expf(a));
  g[i] = -__expf(A_log[h]) * sp;
  float bb = ab[(size_t)bt * kABN + 16 + h];
  beta[i] = 2.f * fsig(bb);
}

// ---------------- chunked gated delta rule (WY form) ----------------
// grid = (32 b*h, kNS val-splits).  K-side work (A, (I+A)^-1, M) is shared
// math duplicated per split; U/V'/o/S are sliced (kSL=32 val channels).
// Q/K/V chunks double-buffered: next chunk's TDM issued before compute.
static constexpr int kDeltaLds = 218624;

__global__ __launch_bounds__(256) void delta_chunk_kernel(
    const bf16_t* __restrict__ qn, const bf16_t* __restrict__ kn,
    const bf16_t* __restrict__ vn, const float* __restrict__ g,
    const float* __restrict__ beta, float* __restrict__ o) {
  extern __shared__ char smem[];
  char* pp = smem;
  float*  S    = (float*)pp;   pp += 128 * kSL * 4;   // f32 state [keych][valsl]
  bf16_t* St   = (bf16_t*)pp;  pp += kSL * 128 * 2;   // S^T [valsl][keych]
  bf16_t* Qc0  = (bf16_t*)pp;  pp += 64 * 128 * 2;    // chunk double buffers
  bf16_t* Qc1  = (bf16_t*)pp;  pp += 64 * 128 * 2;
  bf16_t* Kc0  = (bf16_t*)pp;  pp += 64 * 128 * 2;
  bf16_t* Kc1  = (bf16_t*)pp;  pp += 64 * 128 * 2;
  bf16_t* Vc0  = (bf16_t*)pp;  pp += 64 * kSL * 2;
  bf16_t* Vc1  = (bf16_t*)pp;  pp += 64 * kSL * 2;
  bf16_t* KgT  = (bf16_t*)pp;  pp += 128 * 64 * 2;    // (K*beta*expG)^T
  bf16_t* KdT  = (bf16_t*)pp;  pp += 128 * 64 * 2;    // (K*exp(Gc-G))^T
  bf16_t* VbT  = (bf16_t*)pp;  pp += kSL * 64 * 2;    // (V*beta)^T
  bf16_t* UcT  = (bf16_t*)pp;  pp += kSL * 64 * 2;
  bf16_t* Wc   = (bf16_t*)pp;  pp += 64 * 128 * 2;
  bf16_t* VnT  = (bf16_t*)pp;  pp += kSL * 64 * 2;
  bf16_t* Rb0  = (bf16_t*)pp;  pp += 64 * 64 * 2;     // doubling ping-pong
  bf16_t* Rb1  = (bf16_t*)pp;  pp += 64 * 64 * 2;
  bf16_t* Xb0  = (bf16_t*)pp;  pp += 64 * 64 * 2;
  bf16_t* Xb1  = (bf16_t*)pp;  pp += 64 * 64 * 2;
  bf16_t* XT0  = (bf16_t*)pp;  pp += 64 * 64 * 2;
  bf16_t* XT1  = (bf16_t*)pp;  pp += 64 * 64 * 2;
  bf16_t* Mbf  = (bf16_t*)pp;  pp += 64 * 64 * 2;
  float*  G    = (float*)pp;   pp += 64 * 4;
  float*  sKg  = (float*)pp;   pp += 64 * 4;
  float*  sVb  = (float*)pp;   pp += 64 * 4;
  float*  sKd  = (float*)pp;   pp += 64 * 4;
  float*  sQg  = (float*)pp;   pp += 64 * 4;
  float*  sEgc = (float*)pp;

  bf16_t* Qc[2] = { Qc0, Qc1 };
  bf16_t* Kc[2] = { Kc0, Kc1 };
  bf16_t* Vc[2] = { Vc0, Vc1 };
  bf16_t* Rb[2] = { Rb0, Rb1 };
  bf16_t* Xb[2] = { Xb0, Xb1 };
  bf16_t* XT[2] = { XT0, XT1 };

  int tid = threadIdx.x;
  int wave = tid >> 5;
  int lane = tid & 31;
  int bh = blockIdx.x;
  int sl = blockIdx.y;             // val slice
  int b = bh >> 4;
  int h = bh & 15;
  size_t headBase = (size_t)b * kT * kHID + (size_t)h * kD;

  for (int i = tid; i < 128 * kSL; i += 256) { S[i] = 0.f; St[i] = (bf16_t)0.f; }

  auto stageChunk = [&](int ci2, int buf) {
    size_t gb = headBase + (size_t)(ci2 * kCH) * kHID;
    if (wave == 0)      tdm_load_2d(qn + gb, lds_off(Qc[buf]), 64, 128, kHID);
    else if (wave == 1) tdm_load_2d(kn + gb, lds_off(Kc[buf]), 64, 128, kHID);
    else if (wave == 2) tdm_load_2d(vn + gb + sl * kSL, lds_off(Vc[buf]), 64, kSL, kHID);
  };

  stageChunk(0, 0);
  __builtin_amdgcn_s_wait_tensorcnt(0);
  __syncthreads();

  const int NT = kT / kCH;
  for (int ci = 0; ci < NT; ++ci) {
    int cb = ci & 1;
    if (ci + 1 < NT) stageChunk(ci + 1, cb ^ 1);   // async prefetch next chunk

    if (tid < 64) G[tid] = g[((size_t)b * kT + ci * kCH + tid) * kNH + h];
    __syncthreads();
    if (tid == 0) {
      float acc = 0.f;
      for (int i = 0; i < 64; ++i) { acc += G[i]; G[i] = acc; }
    }
    __syncthreads();
    if (tid < 64) {
      float bt_ = beta[((size_t)b * kT + ci * kCH + tid) * kNH + h];
      float Gc = G[63];
      float eg = fexp(G[tid]);
      sQg[tid] = eg;
      sKg[tid] = bt_ * eg;
      sVb[tid] = bt_;
      sKd[tid] = fexp(Gc - G[tid]);
      if (tid == 0) sEgc[0] = fexp(Gc);
    }
    __syncthreads();

    // ---- scaled transposed operands: KgT/KdT (full K), VbT (slice) ----
    {
      int c = tid >> 1;
      int t0 = (tid & 1) * 32;
#pragma unroll
      for (int blk = 0; blk < 4; ++blk) {
        unsigned wg[4], wd[4];
#pragma unroll
        for (int j = 0; j < 4; ++j) {
          int t = t0 + blk * 8 + j * 2;
          float k0v = (float)Kc[cb][t * 128 + c];
          float k1v = (float)Kc[cb][(t + 1) * 128 + c];
          wg[j] = (unsigned)bf_bits((bf16_t)(k0v * sKg[t])) |
                  ((unsigned)bf_bits((bf16_t)(k1v * sKg[t + 1])) << 16);
          wd[j] = (unsigned)bf_bits((bf16_t)(k0v * sKd[t])) |
                  ((unsigned)bf_bits((bf16_t)(k1v * sKd[t + 1])) << 16);
        }
        int4 p0; p0.x = (int)wg[0]; p0.y = (int)wg[1]; p0.z = (int)wg[2]; p0.w = (int)wg[3];
        int4 p1; p1.x = (int)wd[0]; p1.y = (int)wd[1]; p1.z = (int)wd[2]; p1.w = (int)wd[3];
        *(int4*)(KgT + c * 64 + t0 + blk * 8) = p0;
        *(int4*)(KdT + c * 64 + t0 + blk * 8) = p1;
      }
    }
    if (tid < 2 * kSL) {
      int c = tid >> 1;
      int t0 = (tid & 1) * 32;
#pragma unroll
      for (int blk = 0; blk < 4; ++blk) {
        unsigned wb[4];
#pragma unroll
        for (int j = 0; j < 4; ++j) {
          int t = t0 + blk * 8 + j * 2;
          float v0v = (float)Vc[cb][t * kSL + c];
          float v1v = (float)Vc[cb][(t + 1) * kSL + c];
          wb[j] = (unsigned)bf_bits((bf16_t)(v0v * sVb[t])) |
                  ((unsigned)bf_bits((bf16_t)(v1v * sVb[t + 1])) << 16);
        }
        int4 p2; p2.x = (int)wb[0]; p2.y = (int)wb[1]; p2.z = (int)wb[2]; p2.w = (int)wb[3];
        *(int4*)(VbT + c * 64 + t0 + blk * 8) = p2;
      }
    }
    __syncthreads();

    // ---- X = -A (strictly lower), fused epilogue; R = I + X ----
    for (int t = wave; t < 16; t += 8) {
      int tr = (t >> 2) * 16, tc = (t & 3) * 16;
      v8f acc = {};
#pragma unroll
      for (int k2 = 0; k2 < 128; k2 += 32) {
        v16bf a = ldA_bf(Kc[cb], 128, tr, k2);
        v16bf bb_ = ldBT_bf(Kc[cb], 128, k2, tc);
        acc = WMMA_BF16(a, bb_, acc);
      }
      int j = tc + (lane & 15);
      int rb = tr + ((lane >> 4) << 3);
      v8f xv;
#pragma unroll
      for (int v = 0; v < 8; ++v) {
        int i = rb + v;
        float val = (j < i) ? -(acc[v] * sVb[i] * fexp(G[i] - G[j])) : 0.f;
        xv[v] = val;
        Xb[0][i * 64 + j] = (bf16_t)val;
        Rb[0][i * 64 + j] = (bf16_t)(val + ((i == j) ? 1.f : 0.f));
      }
      stC_bf16_T(XT[0], 64, tr, tc, xv);
    }
    __syncthreads();

    // ---- T = (I+A)^-1 by nilpotent doubling ----
    int cur = 0;
#pragma unroll 1
    for (int it = 0; it < 5; ++it) {
      int nxt = cur ^ 1;
      for (int t = wave; t < 16; t += 8) {
        int tr = (t >> 2) * 16, tc = (t & 3) * 16;
        v8f acc = ldC_bf16(Rb[cur], 64, tr, tc);
        v8f acc2 = {};
#pragma unroll
        for (int k2 = 0; k2 < 64; k2 += 32) {
          v16bf bB = ldA_bf(XT[cur], 64, tc, k2);
          v16bf aR = ldA_bf(Rb[cur], 64, tr, k2);
          v16bf aX = ldA_bf(Xb[cur], 64, tr, k2);
          acc  = WMMA_BF16(aR, bB, acc);
          acc2 = WMMA_BF16(aX, bB, acc2);
        }
        stC_bf16(Rb[nxt], 64, tr, tc, acc);
        stC_bf16(Xb[nxt], 64, tr, tc, acc2);
        stC_bf16_T(XT[nxt], 64, tr, tc, acc2);
      }
      __syncthreads();
      cur = nxt;
    }
    bf16_t* Rfin = Rb[cur];

    // ---- W = T @ Kg (full 128) ; U^T = (T @ Vb)^T (slice) ----
    for (int t = wave; t < 32; t += 8) {
      int tr = (t >> 3) * 16, tc = (t & 7) * 16;
      v8f accw = {};
#pragma unroll
      for (int k2 = 0; k2 < 64; k2 += 32) {
        v16bf ra = ldA_bf(Rfin, 64, tr, k2);
        v16bf kb = ldA_bf(KgT, 64, tc, k2);
        accw = WMMA_BF16(ra, kb, accw);
      }
      stC_bf16(Wc, 128, tr, tc, accw);
    }
    for (int t = wave; t < 4 * (kSL / 16); t += 8) {
      int tr = (t >> 1) * 16, tc = (t & 1) * 16;
      v8f accu = {};
#pragma unroll
      for (int k2 = 0; k2 < 64; k2 += 32) {
        v16bf ra = ldA_bf(Rfin, 64, tr, k2);
        v16bf vb = ldA_bf(VbT, 64, tc, k2);
        accu = WMMA_BF16(ra, vb, accu);
      }
      stC_bf16_T(UcT, 64, tr, tc, accu);
    }
    __syncthreads();

    // ---- V'^T = (U - W @ S)^T ----
    for (int t = wave; t < 4 * (kSL / 16); t += 8) {
      int tr = (t >> 1) * 16, tc = (t & 1) * 16;
      v8f acc = {};
#pragma unroll
      for (int k2 = 0; k2 < 128; k2 += 32) {
        v16bf wa = ldA_bf(Wc, 128, tr, k2);
        v16bf sb = ldA_bf(St, 128, tc, k2);
        acc = WMMA_BF16(wa, sb, acc);
      }
      int c = tc + (lane & 15);
      int rb = tr + ((lane >> 4) << 3);
      v8f tmp;
#pragma unroll
      for (int v = 0; v < 8; ++v) tmp[v] = (float)UcT[c * 64 + rb + v] - acc[v];
      stC_bf16_T(VnT, 64, tr, tc, tmp);
    }
    __syncthreads();

    // ---- M = mask(Q @ K^T) * exp(G_i - G_j), j <= i ----
    for (int t = wave; t < 16; t += 8) {
      int tr = (t >> 2) * 16, tc = (t & 3) * 16;
      v8f acc = {};
#pragma unroll
      for (int k2 = 0; k2 < 128; k2 += 32) {
        v16bf qa = ldA_bf(Qc[cb], 128, tr, k2);
        v16bf kb = ldBT_bf(Kc[cb], 128, k2, tc);
        acc = WMMA_BF16(qa, kb, acc);
      }
      int j = tc + (lane & 15);
      int rb = tr + ((lane >> 4) << 3);
#pragma unroll
      for (int v = 0; v < 8; ++v) {
        int i = rb + v;
        Mbf[i * 64 + j] = (bf16_t)((j <= i) ? acc[v] * fexp(G[i] - G[j]) : 0.f);
      }
    }
    __syncthreads();

    // ---- o_slice = (Q*expG) @ S + M @ V' -> global ----
    for (int t = wave; t < 4 * (kSL / 16); t += 8) {
      int tr = (t >> 1) * 16, tc = (t & 1) * 16;
      v8f acc = {};
#pragma unroll
      for (int k2 = 0; k2 < 128; k2 += 32) {
        v16bf qa = ldA_bf_rowscale(Qc[cb], 128, tr, k2, sQg);
        v16bf sb = ldA_bf(St, 128, tc, k2);
        acc = WMMA_BF16(qa, sb, acc);
      }
#pragma unroll
      for (int k2 = 0; k2 < 64; k2 += 32) {
        v16bf ma = ldA_bf(Mbf, 64, tr, k2);
        v16bf vb = ldA_bf(VnT, 64, tc, k2);
        acc = WMMA_BF16(ma, vb, acc);
      }
      int c = tc + (lane & 15);
      int rb = tr + ((lane >> 4) << 3);
#pragma unroll
      for (int v = 0; v < 8; ++v) {
        o[headBase + (size_t)(ci * kCH + rb + v) * kHID + sl * kSL + c] = acc[v];
      }
    }
    __syncthreads();

    // ---- S = exp(Gc)*S + Kd^T @ V' ; refresh S (f32) and S^T (bf16) ----
    {
      float egc = sEgc[0];
      for (int t = wave; t < 8 * (kSL / 16); t += 8) {
        int tr = (t >> 1) * 16, tc = (t & 1) * 16;
        v8f acc = ldC_f32_scaled(S, kSL, tr, tc, egc);
#pragma unroll
        for (int k2 = 0; k2 < 64; k2 += 32) {
          v16bf ka = ldA_bf(KdT, 64, tr, k2);
          v16bf vb = ldA_bf(VnT, 64, tc, k2);
          acc = WMMA_BF16(ka, vb, acc);
        }
        stC_f32(S, kSL, tr, tc, acc);
        stC_bf16_T(St, 128, tr, tc, acc);
      }
    }
    __builtin_amdgcn_s_wait_tensorcnt(0);   // next chunk's TDM done
    __syncthreads();
  }
}

// ---------------- gate SiLU + RMS norm -> bf16 ----------------
__global__ __launch_bounds__(256) void out_norm_kernel(const float* __restrict__ o,
                                                       const float* __restrict__ ab,
                                                       const float* __restrict__ onw,
                                                       bf16_t* __restrict__ ob) {
  int gw = (blockIdx.x * blockDim.x + threadIdx.x) >> 5;
  int lane = threadIdx.x & 31;
  if (gw >= kBT * kNH) return;
  int bt = gw >> 4;
  int h = gw & 15;
  float y[4];
  float ss = 0.f;
#pragma unroll
  for (int i = 0; i < 4; ++i) {
    int c = lane + 32 * i;
    float ov = o[(size_t)gw * kD + c];
    float gt = ab[(size_t)bt * kABN + 32 + h * kD + c];
    float yy = ov * gt * fsig(gt);
    y[i] = yy;
    ss += yy * yy;
  }
#pragma unroll
  for (int off = 16; off; off >>= 1) ss += __shfl_xor(ss, off);
  float r = rsqrtf(ss * (1.f / 128.f) + 1e-5f);
#pragma unroll
  for (int i = 0; i < 4; ++i) {
    int c = lane + 32 * i;
    ob[(size_t)gw * kD + c] = (bf16_t)(y[i] * r * onw[c]);
  }
}

// ---------------- host launcher ----------------
extern "C" void kernel_launch(void* const* d_in, const int* in_sizes, int n_in,
                              void* d_out, int out_size, void* d_ws, size_t ws_size,
                              hipStream_t stream) {
  (void)in_sizes; (void)n_in; (void)out_size; (void)ws_size;
  const float* hid   = (const float*)d_in[0];
  const float* Wqkv  = (const float*)d_in[1];
  const float* Wab   = (const float*)d_in[2];
  const float* convw = (const float*)d_in[3];
  const float* A_log = (const float*)d_in[4];
  const float* dtb   = (const float*)d_in[5];
  const float* onw   = (const float*)d_in[6];
  const float* Wo    = (const float*)d_in[7];
  float* out = (float*)d_out;

  char* ws = (char*)d_ws;
  size_t off = 0;
  auto alloc = [&](size_t bytes) -> char* {
    char* p = ws + off;
    off += (bytes + 255) & ~(size_t)255;
    return p;
  };

  bf16_t* hidb  = (bf16_t*)alloc((size_t)kBT * kHID * 2);
  bf16_t* wqkvb = (bf16_t*)alloc((size_t)kHID * kQKVN * 2);
  bf16_t* wabb  = (bf16_t*)alloc((size_t)kHID * kABN * 2);
  bf16_t* wob   = (bf16_t*)alloc((size_t)kHID * kHID * 2);
  float*  qkvf  = (float*)alloc((size_t)kBT * kQKVN * 4);
  float*  abf   = (float*)alloc((size_t)kBT * kABN * 4);
  bf16_t* qn    = (bf16_t*)alloc((size_t)kBT * kHID * 2);
  bf16_t* kn    = (bf16_t*)alloc((size_t)kBT * kHID * 2);
  bf16_t* vn    = (bf16_t*)alloc((size_t)kBT * kHID * 2);
  float*  gbuf  = (float*)alloc((size_t)kBT * kNH * 4);
  float*  bbuf  = (float*)alloc((size_t)kBT * kNH * 4);
  float*  obuf  = (float*)alloc((size_t)kBT * kHID * 4);
  bf16_t* ob    = (bf16_t*)alloc((size_t)kBT * kHID * 2);

  auto cvt = [&](const float* src, bf16_t* dst, size_t n) {
    int blocks = (int)((n + 255) / 256);
    cvt_bf16_kernel<<<blocks, 256, 0, stream>>>(src, dst, n);
  };
  cvt(hid,  hidb,  (size_t)kBT * kHID);
  cvt(Wqkv, wqkvb, (size_t)kHID * kQKVN);
  cvt(Wab,  wabb,  (size_t)kHID * kABN);
  cvt(Wo,   wob,   (size_t)kHID * kHID);

  gemm_bf16_kernel<<<dim3(kQKVN / 64, kBT / 64), 256, 0, stream>>>(
      hidb, wqkvb, qkvf, kBT, kQKVN, kHID);
  gemm_bf16_kernel<<<dim3((kABN + 63) / 64, kBT / 64), 256, 0, stream>>>(
      hidb, wabb, abf, kBT, kABN, kHID);

  conv_silu_prep_kernel<<<(kBT * kNH * 32) / 256, 256, 0, stream>>>(
      qkvf, convw, qn, kn, vn);

  gbeta_kernel<<<(kBT * kNH + 255) / 256, 256, 0, stream>>>(abf, A_log, dtb, gbuf, bbuf);

  (void)hipFuncSetAttribute(reinterpret_cast<const void*>(delta_chunk_kernel),
                            hipFuncAttributeMaxDynamicSharedMemorySize, kDeltaLds);
  delta_chunk_kernel<<<dim3(kB * kNH, kNS), 256, kDeltaLds, stream>>>(
      qn, kn, vn, gbuf, bbuf, obuf);

  out_norm_kernel<<<(kBT * kNH * 32) / 256, 256, 0, stream>>>(obuf, abf, onw, ob);

  gemm_bf16_kernel<<<dim3(kHID / 64, kBT / 64), 256, 0, stream>>>(
      ob, wob, out, kBT, kHID, kHID);
}
